// SAGEEdgeClassifier_42597485641878
// MI455X (gfx1250) — matrix operand order repaired and checked by reading, not compiled, not run on hardware
//
#include <hip/hip_runtime.h>
#include <hip/hip_bf16.h>

#define N_NODES 100000
#define N_EDGES 600000
#define IN_DIM 128
#define EDGE_DIM 32
#define HIDDEN 128
#define NUM_CLASSES 2

typedef __attribute__((ext_vector_type(16))) _Float16 v16h;
typedef __attribute__((ext_vector_type(8)))  _Float16 v8h;
typedef __attribute__((ext_vector_type(8)))  float    v8f;

// ---------------------------------------------------------------------------
// WMMA fragment loaders (wave32, 16x16x32 f16, ISA 7.12.2 layouts)
// A: lane L holds row (L&15); K = {0..7, 16..23} (lanes 0-15) or +8 (lanes 16-31)
// B: lane L holds col (L&15); K = kstep+{0..7, 16..23} (lanes 0-15) or +8 (lanes 16-31)
// ---------------------------------------------------------------------------
__device__ __forceinline__ v16h load_a_frag(const _Float16* A, int lane, int kstep, int rowStride) {
    const int r  = lane & 15;
    const int k0 = kstep + ((lane >> 4) << 3);
    v8h lo = *(const v8h*)(A + r * rowStride + k0);
    v8h hi = *(const v8h*)(A + r * rowStride + k0 + 16);
    v16h a;
#pragma unroll
    for (int i = 0; i < 8; ++i) { a[i] = lo[i]; a[i + 8] = hi[i]; }
    return a;
}

__device__ __forceinline__ v16h load_b_frag(const _Float16* Bt, int lane, int kstep, int colBase, int K) {
    const int n  = colBase + (lane & 15);
    const int kb = kstep + ((lane >> 4) << 3);
    const v8h* p = (const v8h*)(Bt + (size_t)n * K + kb);
    v8h lo = p[0], hi = p[2];
    v16h b;
#pragma unroll
    for (int i = 0; i < 8; ++i) { b[i] = lo[i]; b[i + 8] = hi[i]; }
    return b;
}

// ---------------------------------------------------------------------------
// Weight prep: transpose [K,N] -> [N,K], f32 or f16 output
// ---------------------------------------------------------------------------
__global__ void transpose_f32_kernel(const float* __restrict__ W, float* __restrict__ Wt, int K, int Nc) {
    int i = blockIdx.x * 256 + threadIdx.x;
    if (i < K * Nc) { int k = i / Nc, n = i % Nc; Wt[n * K + k] = W[i]; }
}

__global__ void transpose_f16_kernel(const float* __restrict__ W, _Float16* __restrict__ Wt, int K, int Nc) {
    int i = blockIdx.x * 256 + threadIdx.x;
    if (i < K * Nc) { int k = i / Nc, n = i % Nc; Wt[n * K + k] = (_Float16)W[i]; }
}

// ---------------------------------------------------------------------------
// Edge message + scatter:  agg[dst] += feat[src] + edge_attr@We + be
// 256 threads = 2 edges x 128 features. Scatter via global f32 atomics (L2).
// ---------------------------------------------------------------------------
__global__ __launch_bounds__(256)
void sage_edge_kernel(const int* __restrict__ srcIdx, const int* __restrict__ dstIdx,
                      const float* __restrict__ EA, const float* __restrict__ Wet,
                      const float* __restrict__ be, const float* __restrict__ Feat,
                      float* __restrict__ agg, float* __restrict__ deg, int countDeg) {
    __shared__ float ea[2][EDGE_DIM];
    const int t  = threadIdx.x;
    const int eL = t >> 7;
    const int f  = t & 127;
    const long long e = (long long)blockIdx.x * 2 + eL;
    if (t < 64) {
        int r = t >> 5, k = t & 31;
        ea[r][k] = EA[((long long)blockIdx.x * 2 + r) * EDGE_DIM + k];
    }
    __syncthreads();
    float proj = 0.0f;
#pragma unroll
    for (int k = 0; k < EDGE_DIM; ++k) proj += ea[eL][k] * Wet[f * EDGE_DIM + k];
    const int s = srcIdx[e];
    const int d = dstIdx[e];
    const float val = Feat[(long long)s * 128 + f] + proj + be[f];
    atomicAdd(&agg[(long long)d * 128 + f], val);
    if (countDeg && f == 0) atomicAdd(&deg[d], 1.0f);
}

// ---------------------------------------------------------------------------
// Node update: H = relu( (agg/max(deg,1)) @ Wl + Feat @ Wr + bl + br )
// Block = 256 thr = 8 waves; 16 nodes x 128 cols per block; 8 WMMA per wave.
// ---------------------------------------------------------------------------
__global__ __launch_bounds__(256)
void sage_node_kernel(const float* __restrict__ agg, const float* __restrict__ deg,
                      const float* __restrict__ Feat,
                      const _Float16* __restrict__ Wlt, const _Float16* __restrict__ Wrt,
                      const float* __restrict__ bl, const float* __restrict__ br,
                      float* __restrict__ Hout) {
    __shared__ __align__(16) _Float16 Am[16 * 128];
    __shared__ __align__(16) _Float16 Ax[16 * 128];
    const int t  = threadIdx.x;
    const int nb = blockIdx.x * 16;
    for (int idx = t; idx < 16 * 128; idx += 256) {
        const int row = idx >> 7, col = idx & 127;
        const int n = nb + row;
        float dg = deg[n]; dg = dg > 1.0f ? dg : 1.0f;
        Am[idx] = (_Float16)(agg[(long long)n * 128 + col] / dg);
        Ax[idx] = (_Float16)(Feat[(long long)n * 128 + col]);
    }
    __syncthreads();
    const int wave = t >> 5, lane = t & 31;
    const int colBase = wave * 16;
    v8f c = {};
#pragma unroll
    for (int kstep = 0; kstep < 128; kstep += 32) {
        v16h a = load_a_frag(Am, lane, kstep, 128);
        v16h b = load_b_frag(Wlt, lane, kstep, colBase, 128);
        c = __builtin_amdgcn_wmma_f32_16x16x32_f16(false, a, false, b, (short)0, c, false, false);
        a = load_a_frag(Ax, lane, kstep, 128);
        b = load_b_frag(Wrt, lane, kstep, colBase, 128);
        c = __builtin_amdgcn_wmma_f32_16x16x32_f16(false, a, false, b, (short)0, c, false, false);
    }
    const int col   = colBase + (lane & 15);
    const float bias = bl[col] + br[col];
    const int rbase = nb + ((lane >> 4) << 3);
#pragma unroll
    for (int v = 0; v < 8; ++v) {
        float r = c[v] + bias;
        r = r > 0.0f ? r : 0.0f;
        Hout[(long long)(rbase + v) * 128 + col] = r;
    }
}

// ---------------------------------------------------------------------------
// Edge predictor: he = [h2[src], h2[dst]];  out = relu(he@Wp1+bp1) @ Wp2 + bp2
// Block = 256 thr; 16 edges per block; K=256 GEMM on WMMA, final N=2 on VALU.
// ---------------------------------------------------------------------------
__global__ __launch_bounds__(256)
void edge_pred_kernel(const int* __restrict__ srcIdx, const int* __restrict__ dstIdx,
                      const float* __restrict__ H2,
                      const _Float16* __restrict__ Wp1t, const float* __restrict__ bp1,
                      const float* __restrict__ Wp2, const float* __restrict__ bp2,
                      float* __restrict__ out) {
    __shared__ __align__(16) _Float16 He[16 * 256];
    __shared__ __align__(16) _Float16 Rm[16 * 128];
    const int t = threadIdx.x;
    const long long ebase = (long long)blockIdx.x * 16;
    for (int idx = t; idx < 16 * 256; idx += 256) {
        const int row = idx >> 8, col = idx & 255;
        const long long e = ebase + row;
        const int node = (col < 128) ? srcIdx[e] : dstIdx[e];
        He[idx] = (_Float16)H2[(long long)node * 128 + (col & 127)];
    }
    __syncthreads();
    const int wave = t >> 5, lane = t & 31;
    const int colBase = wave * 16;
    v8f c = {};
#pragma unroll
    for (int kstep = 0; kstep < 256; kstep += 32) {
        v16h a = load_a_frag(He, lane, kstep, 256);
        v16h b = load_b_frag(Wp1t, lane, kstep, colBase, 256);
        c = __builtin_amdgcn_wmma_f32_16x16x32_f16(false, a, false, b, (short)0, c, false, false);
    }
    const int col   = colBase + (lane & 15);
    const float bias = bp1[col];
    const int rbase = (lane >> 4) << 3;
#pragma unroll
    for (int v = 0; v < 8; ++v) {
        float r = c[v] + bias;
        r = r > 0.0f ? r : 0.0f;
        Rm[(rbase + v) * 128 + col] = (_Float16)r;
    }
    __syncthreads();
    if (t < 32) {
        const int eL = t >> 1, cls = t & 1;
        float acc = 0.0f;
#pragma unroll 8
        for (int k = 0; k < 128; ++k) acc += (float)Rm[eL * 128 + k] * Wp2[k * NUM_CLASSES + cls];
        out[(ebase + eL) * NUM_CLASSES + cls] = acc + bp2[cls];
    }
}

// ---------------------------------------------------------------------------
extern "C" void kernel_launch(void* const* d_in, const int* in_sizes, int n_in,
                              void* d_out, int out_size, void* d_ws, size_t ws_size,
                              hipStream_t stream) {
    (void)in_sizes; (void)n_in; (void)out_size; (void)ws_size;
    const float* x   = (const float*)d_in[0];
    const int*   ei  = (const int*)d_in[1];
    const float* ea  = (const float*)d_in[2];
    const float* We1 = (const float*)d_in[3];
    const float* be1 = (const float*)d_in[4];
    const float* Wl1 = (const float*)d_in[5];
    const float* bl1 = (const float*)d_in[6];
    const float* Wr1 = (const float*)d_in[7];
    const float* br1 = (const float*)d_in[8];
    const float* We2 = (const float*)d_in[9];
    const float* be2 = (const float*)d_in[10];
    const float* Wl2 = (const float*)d_in[11];
    const float* bl2 = (const float*)d_in[12];
    const float* Wr2 = (const float*)d_in[13];
    const float* br2 = (const float*)d_in[14];
    const float* Wp1 = (const float*)d_in[15];
    const float* bp1 = (const float*)d_in[16];
    const float* Wp2 = (const float*)d_in[17];
    const float* bp2 = (const float*)d_in[18];
    const int* srcI = ei;
    const int* dstI = ei + N_EDGES;

    // Workspace carve (256B aligned)
    char* p = (char*)d_ws;
    auto take = [&](size_t bytes) { char* r = p; p += (bytes + 255) & ~(size_t)255; return r; };
    float*    agg   = (float*)take((size_t)N_NODES * 128 * sizeof(float));
    float*    deg   = (float*)take((size_t)N_NODES * sizeof(float));
    float*    h1    = (float*)take((size_t)N_NODES * 128 * sizeof(float));
    float*    h2    = (float*)take((size_t)N_NODES * 128 * sizeof(float));
    float*    Wet1  = (float*)take(128 * 32 * sizeof(float));
    float*    Wet2  = (float*)take(128 * 32 * sizeof(float));
    _Float16* Wl1t  = (_Float16*)take(128 * 128 * sizeof(_Float16));
    _Float16* Wr1t  = (_Float16*)take(128 * 128 * sizeof(_Float16));
    _Float16* Wl2t  = (_Float16*)take(128 * 128 * sizeof(_Float16));
    _Float16* Wr2t  = (_Float16*)take(128 * 128 * sizeof(_Float16));
    _Float16* Wp1t  = (_Float16*)take(256 * 128 * sizeof(_Float16));

    // Weight prep
    transpose_f32_kernel<<<(32 * 128 + 255) / 256, 256, 0, stream>>>(We1, Wet1, 32, 128);
    transpose_f32_kernel<<<(32 * 128 + 255) / 256, 256, 0, stream>>>(We2, Wet2, 32, 128);
    transpose_f16_kernel<<<(128 * 128 + 255) / 256, 256, 0, stream>>>(Wl1, Wl1t, 128, 128);
    transpose_f16_kernel<<<(128 * 128 + 255) / 256, 256, 0, stream>>>(Wr1, Wr1t, 128, 128);
    transpose_f16_kernel<<<(128 * 128 + 255) / 256, 256, 0, stream>>>(Wl2, Wl2t, 128, 128);
    transpose_f16_kernel<<<(128 * 128 + 255) / 256, 256, 0, stream>>>(Wr2, Wr2t, 128, 128);
    transpose_f16_kernel<<<(256 * 128 + 255) / 256, 256, 0, stream>>>(Wp1, Wp1t, 256, 128);

    // Layer 1
    hipMemsetAsync(agg, 0, (size_t)N_NODES * 128 * sizeof(float), stream);
    hipMemsetAsync(deg, 0, (size_t)N_NODES * sizeof(float), stream);
    sage_edge_kernel<<<N_EDGES / 2, 256, 0, stream>>>(srcI, dstI, ea, Wet1, be1, x, agg, deg, 1);
    sage_node_kernel<<<N_NODES / 16, 256, 0, stream>>>(agg, deg, x, Wl1t, Wr1t, bl1, br1, h1);

    // Layer 2 (reuse agg; deg identical across layers)
    hipMemsetAsync(agg, 0, (size_t)N_NODES * 128 * sizeof(float), stream);
    sage_edge_kernel<<<N_EDGES / 2, 256, 0, stream>>>(srcI, dstI, ea, Wet2, be2, h1, agg, deg, 0);
    sage_node_kernel<<<N_NODES / 16, 256, 0, stream>>>(agg, deg, h1, Wl2t, Wr2t, bl2, br2, h2);

    // Edge predictor
    edge_pred_kernel<<<N_EDGES / 16, 256, 0, stream>>>(srcI, dstI, h2, Wp1t, bp1, Wp2, bp2, (float*)d_out);
}
